// Block_75720273428700
// MI455X (gfx1250) — matrix operand order repaired and checked by reading, not compiled, not run on hardware
//
#include <hip/hip_runtime.h>

typedef __attribute__((ext_vector_type(16))) _Float16 v16h;
typedef __attribute__((ext_vector_type(8)))  _Float16 v8h;
typedef __attribute__((ext_vector_type(8)))  float    v8f;

#define N_EMBD 768
#define NHEAD  12
#define HDIM   64
#define HIDDEN 3072
#define BATCH  4
#define SEQ    2048
#define MROWS  (BATCH*SEQ)   /* 8192 */
#define QKV3   (3*N_EMBD)    /* 2304 */

// ---------------------------------------------------------------------------
// Weight convert + transpose: W[K][N] f32 -> Wt[N][K] f16 (K-major rows so
// every WMMA B-fragment is one contiguous 32B load per lane).
// ---------------------------------------------------------------------------
__global__ void wconv_kernel(const float* __restrict__ W, _Float16* __restrict__ Wt,
                             int K, int N) {
    int idx = blockIdx.x * 256 + threadIdx.x;
    if (idx >= K * N) return;
    int k = idx / N, n = idx % N;          // coalesced read over n
    Wt[(size_t)n * K + k] = (_Float16)W[idx];
}

// ---------------------------------------------------------------------------
// LayerNorm: one block per row of 768, f32 in -> f16 out (WMMA A operand).
// ---------------------------------------------------------------------------
__global__ void ln_kernel(const float* __restrict__ x, const float* __restrict__ g,
                          const float* __restrict__ bb, _Float16* __restrict__ out) {
    __shared__ float s1[256], s2[256];
    const int row = blockIdx.x;
    const float* xr = x + (size_t)row * N_EMBD;
    float a = 0.f, b = 0.f;
    for (int i = threadIdx.x; i < N_EMBD; i += 256) { float v = xr[i]; a += v; b += v * v; }
    s1[threadIdx.x] = a; s2[threadIdx.x] = b;
    __syncthreads();
    for (int off = 128; off > 0; off >>= 1) {
        if (threadIdx.x < off) { s1[threadIdx.x] += s1[threadIdx.x + off];
                                 s2[threadIdx.x] += s2[threadIdx.x + off]; }
        __syncthreads();
    }
    const float mu  = s1[0] * (1.0f / N_EMBD);
    const float var = s2[0] * (1.0f / N_EMBD) - mu * mu;
    const float rs  = rsqrtf(var + 1e-5f);
    for (int i = threadIdx.x; i < N_EMBD; i += 256)
        out[(size_t)row * N_EMBD + i] = (_Float16)((xr[i] - mu) * rs * g[i] + bb[i]);
}

// ---------------------------------------------------------------------------
// WMMA helpers
// ---------------------------------------------------------------------------
__device__ __forceinline__ v8f wmma_f16(v16h a, v16h b, v8f c) {
    return __builtin_amdgcn_wmma_f32_16x16x32_f16(false, a, false, b, (short)0, c,
                                                  false, false);
}
__device__ __forceinline__ v16h ld_afrag(const _Float16* p0, const _Float16* p1) {
    v8h lo = *(const v8h*)p0;
    v8h hi = *(const v8h*)p1;
    v16h r;
#pragma unroll
    for (int i = 0; i < 8; ++i) { r[i] = lo[i]; r[8 + i] = hi[i]; }
    return r;
}

// ---------------------------------------------------------------------------
// Wave-tile GEMM: C[M,N] = A[M,K](f16, row-major) x Wt[N,K](f16, K-major).
// Each wave -> one 32x64 tile: 2 A-frags + 4 B-frags + 8 v_wmma per 32-K step
// (B-fragments reused across both A row-groups -> ~21 FLOP/byte from L2;
//  all f16 weights fit in the 192MB L2 so this is the binding path).
// Epilogue: 0 = f32 store, 1 = f32 store + residual, 2 = f16 store,
//           3 = exact-erf GELU then f16 store.
// ---------------------------------------------------------------------------
__global__ void __launch_bounds__(256)
gemm_wmma_kernel(const _Float16* __restrict__ A, const _Float16* __restrict__ Bt,
                 float* __restrict__ Cf, _Float16* __restrict__ Ch,
                 const float* __restrict__ Res,
                 int M, int N, int K, int epilogue) {
    const int wave   = (blockIdx.x * 256 + threadIdx.x) >> 5;
    const int lane   = threadIdx.x & 31;
    const int wavesN = N >> 6;
    const int tileM  = (wave / wavesN) << 5;    // 32 rows per wave
    const int tileN  = (wave % wavesN) << 6;    // 64 cols per wave
    const int g   = lane >> 4;
    const int r16 = lane & 15;

    const _Float16* Arow0 = A + (size_t)(tileM + r16) * K;
    const _Float16* Arow1 = Arow0 + (size_t)16 * K;
    v8f acc[2][4] = {{v8f{}, v8f{}, v8f{}, v8f{}}, {v8f{}, v8f{}, v8f{}, v8f{}}};

    for (int k0 = 0; k0 < K; k0 += 32) {
        v16h a0 = ld_afrag(Arow0 + k0 + 16 * g, Arow0 + k0 + 8 + 16 * g);
        v16h a1 = ld_afrag(Arow1 + k0 + 16 * g, Arow1 + k0 + 8 + 16 * g);
        const _Float16* Bk = Bt + k0 + 16 * g;
        v16h b0 = *(const v16h*)(Bk + (size_t)(tileN +  0 + r16) * K);
        v16h b1 = *(const v16h*)(Bk + (size_t)(tileN + 16 + r16) * K);
        v16h b2 = *(const v16h*)(Bk + (size_t)(tileN + 32 + r16) * K);
        v16h b3 = *(const v16h*)(Bk + (size_t)(tileN + 48 + r16) * K);
        acc[0][0] = wmma_f16(a0, b0, acc[0][0]);
        acc[1][0] = wmma_f16(a1, b0, acc[1][0]);
        acc[0][1] = wmma_f16(a0, b1, acc[0][1]);
        acc[1][1] = wmma_f16(a1, b1, acc[1][1]);
        acc[0][2] = wmma_f16(a0, b2, acc[0][2]);
        acc[1][2] = wmma_f16(a1, b2, acc[1][2]);
        acc[0][3] = wmma_f16(a0, b3, acc[0][3]);
        acc[1][3] = wmma_f16(a1, b3, acc[1][3]);
    }

#pragma unroll
    for (int t = 0; t < 2; ++t) {
#pragma unroll
        for (int sub = 0; sub < 4; ++sub) {
            const int col = tileN + sub * 16 + r16;
#pragma unroll
            for (int r = 0; r < 8; ++r) {
                const int row = tileM + 16 * t + r + 8 * g;   // C layout: M = r + 8*(lane/16)
                const size_t idx = (size_t)row * N + col;
                float v = acc[t][sub][r];
                if (epilogue == 1)      Cf[idx] = v + Res[idx];
                else if (epilogue == 2) Ch[idx] = (_Float16)v;
                else if (epilogue == 3) Ch[idx] = (_Float16)(0.5f * v *
                                            (1.0f + erff(v * 0.70710678118654752f)));
                else                    Cf[idx] = v;
            }
        }
    }
}

// ---------------------------------------------------------------------------
// Flash attention: one wave per (b, h, 16-query tile); online softmax.
// S = Q Kt via WMMA (2 chained per 16-key subtile), P via LDS repack,
// O += P V via WMMA with V staged in LDS.
// ---------------------------------------------------------------------------
__global__ void __launch_bounds__(256)
attn_kernel(const _Float16* __restrict__ qkv, _Float16* __restrict__ yh) {
    __shared__ _Float16 smem[8 * 2560];          // per wave: 512 (P) + 2048 (V) halves
    const int wib  = threadIdx.x >> 5;
    const int lane = threadIdx.x & 31;
    const int wave = blockIdx.x * 8 + wib;
    const int qt = wave % (SEQ / 16);
    const int h  = (wave / (SEQ / 16)) % NHEAD;
    const int b  = wave / ((SEQ / 16) * NHEAD);
    const int qbase = qt * 16;
    const int g = lane >> 4, r16 = lane & 15;

    _Float16* Pb = smem + wib * 2560;
    _Float16* Vb = Pb + 512;

    // Q A-fragments (contraction over d = 0..63, two 32-wide chunks)
    const _Float16* Qrow = qkv + (size_t)(b * SEQ + qbase + r16) * QKV3 + h * HDIM;
    const v16h qa0 = ld_afrag(Qrow + 0  + 16 * g, Qrow + 8 + 16 * g);
    const v16h qa1 = ld_afrag(Qrow + 32 + 16 * g, Qrow + 40 + 16 * g);

    float mrow[8], lrow[8];
    v8f o[4] = {v8f{}, v8f{}, v8f{}, v8f{}};
#pragma unroll
    for (int r = 0; r < 8; ++r) { mrow[r] = -3.0e38f; lrow[r] = 0.f; }

    const int qhi = qbase + 15;
    for (int kb = 0; kb <= qhi; kb += 32) {
        // stage V tile [32 keys x 64] into LDS (one key row per lane)
        {
            int key = kb + lane; if (key > SEQ - 1) key = SEQ - 1;
            const _Float16* Vr = qkv + (size_t)(b * SEQ + key) * QKV3 + 2 * N_EMBD + h * HDIM;
#pragma unroll
            for (int j = 0; j < 8; ++j)
                *(v8h*)(Vb + lane * 64 + j * 8) = *(const v8h*)(Vr + j * 8);
        }
        // S = Q K^T for two 16-key subtiles, scale + causal mask
        float Sf[2][8];
#pragma unroll
        for (int s = 0; s < 2; ++s) {
            int key = kb + 16 * s + r16;
            int keyc = key > SEQ - 1 ? SEQ - 1 : key;
            const _Float16* Kr = qkv + (size_t)(b * SEQ + keyc) * QKV3 + N_EMBD + h * HDIM;
            v16h kb0 = *(const v16h*)(Kr + 16 * g);       // B-frag: K = 16g..16g+15
            v16h kb1 = *(const v16h*)(Kr + 32 + 16 * g);
            v8f sacc = {};
            sacc = wmma_f16(qa0, kb0, sacc);
            sacc = wmma_f16(qa1, kb1, sacc);
            const int kcol = kb + 16 * s + r16;           // C-layout column
#pragma unroll
            for (int r = 0; r < 8; ++r) {
                const int qrow = qbase + r + 8 * g;
                float v = sacc[r] * 0.125f;               // 1/sqrt(64)
                Sf[s][r] = (kcol > qrow) ? -3.0e38f : v;
            }
        }
        // online softmax: row max / rescale / row sum (rows live across 16 lanes)
        float mnew[8], sc[8], rsum[8];
#pragma unroll
        for (int r = 0; r < 8; ++r) {
            float mx = fmaxf(Sf[0][r], Sf[1][r]);
#pragma unroll
            for (int off = 8; off > 0; off >>= 1)
                mx = fmaxf(mx, __shfl_xor(mx, off, 16));
            mnew[r] = fmaxf(mrow[r], mx);
            sc[r]   = expf(mrow[r] - mnew[r]);
            mrow[r] = mnew[r];
            rsum[r] = 0.f;
        }
#pragma unroll
        for (int s = 0; s < 2; ++s)
#pragma unroll
            for (int r = 0; r < 8; ++r) {
                float p = expf(Sf[s][r] - mnew[r]);
                rsum[r] += p;
                Pb[(r + 8 * g) * 32 + s * 16 + r16] = (_Float16)p;  // [qrow][keylocal]
            }
#pragma unroll
        for (int r = 0; r < 8; ++r) {
            float rs = rsum[r];
#pragma unroll
            for (int off = 8; off > 0; off >>= 1)
                rs += __shfl_xor(rs, off, 16);
            lrow[r] = lrow[r] * sc[r] + rs;
#pragma unroll
            for (int sub = 0; sub < 4; ++sub) o[sub][r] *= sc[r];
        }
        asm volatile("s_wait_dscnt 0" ::: "memory");      // LDS RAW across lanes

        // O += P V : A-frag from LDS P, B-frags gathered from LDS V
        v16h pa = ld_afrag(Pb + r16 * 32 + 16 * g, Pb + r16 * 32 + 8 + 16 * g);
#pragma unroll
        for (int sub = 0; sub < 4; ++sub) {
            v16h vb;
            const int d = sub * 16 + r16;
#pragma unroll
            for (int j = 0; j < 16; ++j)
                vb[j] = Vb[(16 * g + j) * 64 + d];        // key = 16g + j
            o[sub] = wmma_f16(pa, vb, o[sub]);
        }
    }

#pragma unroll
    for (int r = 0; r < 8; ++r) {
        const float inv = 1.0f / lrow[r];
        const int row = qbase + r + 8 * g;
#pragma unroll
        for (int sub = 0; sub < 4; ++sub)
            yh[(size_t)(b * SEQ + row) * N_EMBD + h * HDIM + sub * 16 + r16] =
                (_Float16)(o[sub][r] * inv);
    }
}

// ---------------------------------------------------------------------------
extern "C" void kernel_launch(void* const* d_in, const int* in_sizes, int n_in,
                              void* d_out, int out_size, void* d_ws, size_t ws_size,
                              hipStream_t stream) {
    (void)in_sizes; (void)n_in; (void)out_size; (void)ws_size;
    const float* x      = (const float*)d_in[0];
    const float* ln1_g  = (const float*)d_in[1];
    const float* ln1_b  = (const float*)d_in[2];
    const float* W_attn = (const float*)d_in[3];
    const float* W_o    = (const float*)d_in[4];
    const float* ln2_g  = (const float*)d_in[5];
    const float* ln2_b  = (const float*)d_in[6];
    const float* W_fc   = (const float*)d_in[7];
    const float* W_proj = (const float*)d_in[8];
    float* out = (float*)d_out;

    char* ws = (char*)d_ws;
    size_t off = 0;
    auto alloc = [&](size_t bytes) -> void* {
        void* p = ws + off; off = (off + bytes + 255) & ~(size_t)255; return p;
    };
    _Float16* wt_attn = (_Float16*)alloc((size_t)N_EMBD * QKV3   * 2);
    _Float16* wt_o    = (_Float16*)alloc((size_t)N_EMBD * N_EMBD * 2);
    _Float16* wt_fc   = (_Float16*)alloc((size_t)N_EMBD * HIDDEN * 2);
    _Float16* wt_proj = (_Float16*)alloc((size_t)HIDDEN * N_EMBD * 2);
    _Float16* xln_h   = (_Float16*)alloc((size_t)MROWS * N_EMBD * 2);
    _Float16* qkv_h   = (_Float16*)alloc((size_t)MROWS * QKV3   * 2);
    _Float16* y_h     = (_Float16*)alloc((size_t)MROWS * N_EMBD * 2);
    float*    x2      = (float*)   alloc((size_t)MROWS * N_EMBD * 4);
    _Float16* hln_h   = (_Float16*)alloc((size_t)MROWS * N_EMBD * 2);
    _Float16* fc_h    = (_Float16*)alloc((size_t)MROWS * HIDDEN * 2);

    // weights -> f16, transposed to [N][K]
    wconv_kernel<<<(N_EMBD * QKV3   + 255) / 256, 256, 0, stream>>>(W_attn, wt_attn, N_EMBD, QKV3);
    wconv_kernel<<<(N_EMBD * N_EMBD + 255) / 256, 256, 0, stream>>>(W_o,    wt_o,    N_EMBD, N_EMBD);
    wconv_kernel<<<(N_EMBD * HIDDEN + 255) / 256, 256, 0, stream>>>(W_fc,   wt_fc,   N_EMBD, HIDDEN);
    wconv_kernel<<<(HIDDEN * N_EMBD + 255) / 256, 256, 0, stream>>>(W_proj, wt_proj, HIDDEN, N_EMBD);

    // x -> LN1 -> f16
    ln_kernel<<<MROWS, 256, 0, stream>>>(x, ln1_g, ln1_b, xln_h);
    // qkv = ln1(x) @ W_attn  (f16 out)
    gemm_wmma_kernel<<<(MROWS / 32) * (QKV3 / 64) / 8, 256, 0, stream>>>(
        xln_h, wt_attn, nullptr, qkv_h, nullptr, MROWS, QKV3, N_EMBD, 2);
    // causal flash attention -> y (f16)
    attn_kernel<<<BATCH * NHEAD * (SEQ / 16) / 8, 256, 0, stream>>>(qkv_h, y_h);
    // x2 = x + y @ W_o  (f32)
    gemm_wmma_kernel<<<(MROWS / 32) * (N_EMBD / 64) / 8, 256, 0, stream>>>(
        y_h, wt_o, x2, nullptr, x, MROWS, N_EMBD, N_EMBD, 1);
    // LN2 -> f16
    ln_kernel<<<MROWS, 256, 0, stream>>>(x2, ln2_g, ln2_b, hln_h);
    // fc = gelu(ln2 @ W_fc)  (f16, fused exact-erf GELU epilogue)
    gemm_wmma_kernel<<<(MROWS / 32) * (HIDDEN / 64) / 8, 256, 0, stream>>>(
        hln_h, wt_fc, nullptr, fc_h, nullptr, MROWS, HIDDEN, N_EMBD, 3);
    // out = x2 + fc @ W_proj  (f32)
    gemm_wmma_kernel<<<(MROWS / 32) * (N_EMBD / 64) / 8, 256, 0, stream>>>(
        fc_h, wt_proj, out, nullptr, x2, MROWS, N_EMBD, HIDDEN, 1);
}
